// PWLOKANLinear_72284299591943
// MI455X (gfx1250) — matrix operand order repaired and checked by reading, not compiled, not run on hardware
//
#include <hip/hip_runtime.h>
#include <hip/hip_bf16.h>
#include <stdint.h>
#include <stddef.h>

// ---------------------------------------------------------------------------
// PWL-OKAN linear as sparse-expansion GEMM on CDNA5 WMMA, bf16x2 compensated:
//   y = Xa @ A + Xi @ B
//   Xa[b, i*8+s] = (seg(b,i)==s) ? xn(b,i) : 0       (split hi+lo)
//   Xi[b, i*8+s] = (seg(b,i)==s) ? 1      : 0        (exact in bf16)
//   A, B pre-split once into bf16 hi+lo arrays (wsplit_kernel), then staged
//   into LDS with gfx1250 async load-to-LDS (ASYNCcnt) when available.
//   x*w ~= xh*wh + xl*wh + xh*wl   (drop xl*wl ~ 2^-18)
// ---------------------------------------------------------------------------

typedef __bf16 bf16_t;
typedef __bf16 v16bf __attribute__((ext_vector_type(16)));
typedef __bf16 v8bf  __attribute__((ext_vector_type(8)));
typedef float  v8f   __attribute__((ext_vector_type(8)));
typedef int    v4i_vs __attribute__((vector_size(4 * sizeof(int))));

#define IN_F   256
#define OUT_F  512
#define NSEG   8
#define BM     128   // rows per workgroup
#define BN     64    // cols per workgroup
#define KC     32    // expanded-K per step = 4 features * 8 segments
#define LDSP   72    // padded LDS row pitch (bf16 elems) to spread banks

#if __has_builtin(__builtin_amdgcn_global_load_async_to_lds_b128)
#define HAVE_ASYNC_LDS 1
#endif

#define AS1 __attribute__((address_space(1)))
#define AS3 __attribute__((address_space(3)))

static __device__ __forceinline__ v16bf cat16(v8bf lo, v8bf hi) {
    return __builtin_shufflevector(lo, hi, 0, 1, 2, 3, 4, 5, 6, 7,
                                           8, 9, 10, 11, 12, 13, 14, 15);
}

// Stage one 16-byte chunk global->LDS (async if toolchain exposes it).
static __device__ __forceinline__ void stage16(const bf16_t* g, bf16_t* l) {
#if defined(HAVE_ASYNC_LDS)
    AS1 void* gv = (AS1 void*)g;   // addrspacecast generic -> global
    AS3 void* lv = (AS3 void*)l;   // addrspacecast generic -> LDS
    __builtin_amdgcn_global_load_async_to_lds_b128(
        (AS1 v4i_vs*)gv, (AS3 v4i_vs*)lv, 0, 0);
#else
    *(uint4*)l = *(const uint4*)g;
#endif
}

static __device__ __forceinline__ void stage_wait() {
#if defined(HAVE_ASYNC_LDS)
#if __has_builtin(__builtin_amdgcn_s_wait_asynccnt)
    __builtin_amdgcn_s_wait_asynccnt(0);
#else
    asm volatile("s_wait_asynccnt 0x0" ::: "memory");
#endif
#endif
}

// ---------------------------------------------------------------------------
// Kernel 0: one-time weight split  w -> (bf16 hi, bf16 lo).
// ---------------------------------------------------------------------------
__global__ __launch_bounds__(256)
void wsplit_kernel(const float* __restrict__ aW,
                   const float* __restrict__ bW,
                   bf16_t* __restrict__ ah, bf16_t* __restrict__ al,
                   bf16_t* __restrict__ bh, bf16_t* __restrict__ bl) {
    const size_t i = ((size_t)blockIdx.x * 256 + threadIdx.x) * 4;
    float4 a = *(const float4*)(aW + i);
    float4 b = *(const float4*)(bW + i);
    float av[4] = {a.x, a.y, a.z, a.w};
    float bv[4] = {b.x, b.y, b.z, b.w};
    #pragma unroll
    for (int j = 0; j < 4; ++j) {
        bf16_t h = (bf16_t)av[j];
        ah[i + j] = h;
        al[i + j] = (bf16_t)(av[j] - (float)h);
        bf16_t g = (bf16_t)bv[j];
        bh[i + j] = g;
        bl[i + j] = (bf16_t)(bv[j] - (float)g);
    }
}

// ---------------------------------------------------------------------------
// Kernel 1: LayerNorm + segment index.  One block (256 threads) per row.
// ---------------------------------------------------------------------------
__global__ __launch_bounds__(IN_F)
void ln_seg_kernel(const float* __restrict__ x,
                   const float* __restrict__ gamma,
                   const float* __restrict__ beta,
                   bf16_t* __restrict__ xh_out,
                   bf16_t* __restrict__ xl_out,
                   uint8_t* __restrict__ seg_out) {
    __shared__ float s_sum[IN_F];
    __shared__ float s_sq[IN_F];
    const int row = blockIdx.x;
    const int t   = threadIdx.x;

    float v = x[(size_t)row * IN_F + t];
    s_sum[t] = v;
    s_sq[t]  = v * v;
    __syncthreads();
    #pragma unroll
    for (int s = IN_F / 2; s > 0; s >>= 1) {
        if (t < s) { s_sum[t] += s_sum[t + s]; s_sq[t] += s_sq[t + s]; }
        __syncthreads();
    }
    const float mu  = s_sum[0] * (1.0f / IN_F);
    const float var = s_sq[0] * (1.0f / IN_F) - mu * mu;
    const float inv = rsqrtf(var + 1e-5f);
    const float xn  = (v - mu) * inv * gamma[t] + beta[t];

    // seg = clip(trunc((xn - GRID_MIN)/STEP), 0, 7); C cast == trunc
    int seg = (int)((xn + 1.0f) * 4.0f);
    seg = seg < 0 ? 0 : (seg > (NSEG - 1) ? (NSEG - 1) : seg);

    const bf16_t xh = (bf16_t)xn;
    const bf16_t xl = (bf16_t)(xn - (float)xh);
    xh_out[(size_t)row * IN_F + t]  = xh;
    xl_out[(size_t)row * IN_F + t]  = xl;
    seg_out[(size_t)row * IN_F + t] = (uint8_t)seg;
}

// ---------------------------------------------------------------------------
// Kernel 2: expanded GEMM on v_wmma_f32_16x16x32_bf16 (bf16x2 compensated).
// Grid: (OUT_F/BN, rows/BM), 256 threads = 8 waves.
// ---------------------------------------------------------------------------
__global__ __launch_bounds__(256)
void pwl_wmma_kernel(const bf16_t* __restrict__ xh,
                     const bf16_t* __restrict__ xl,
                     const uint8_t* __restrict__ seg,
                     const bf16_t* __restrict__ ah,   // [IN_F*8, OUT_F] hi
                     const bf16_t* __restrict__ al,   // [IN_F*8, OUT_F] lo
                     const bf16_t* __restrict__ bh,
                     const bf16_t* __restrict__ bl,
                     float* __restrict__ y) {         // [rows, OUT_F]
    __shared__ __align__(16) bf16_t lds_ah[KC][LDSP];
    __shared__ __align__(16) bf16_t lds_al[KC][LDSP];
    __shared__ __align__(16) bf16_t lds_bh[KC][LDSP];
    __shared__ __align__(16) bf16_t lds_bl[KC][LDSP];

    const int cbase  = blockIdx.x * BN;
    const int rbase0 = blockIdx.y * BM;
    const int t    = threadIdx.x;
    const int wave = t >> 5;
    const int lane = t & 31;
    const int m    = lane & 15;
    const int half = lane >> 4;
    const int row  = rbase0 + wave * 16 + m;   // A-fragment source row

    v8f acc[4] = {};

    // staging coords: 32 rows x 64 cols bf16, one 16B chunk per array/thread
    const int krow = t >> 3;
    const int kcol = (t & 7) * 8;

    for (int kf = 0; kf < IN_F; kf += 4) {
        const int kglob = kf * NSEG;           // expanded-K base (= i*8)
        const size_t goff = (size_t)(kglob + krow) * OUT_F + cbase + kcol;
        __syncthreads();                       // prior reads done before overwrite
        stage16(ah + goff, &lds_ah[krow][kcol]);
        stage16(al + goff, &lds_al[krow][kcol]);
        stage16(bh + goff, &lds_bh[krow][kcol]);
        stage16(bl + goff, &lds_bl[krow][kcol]);
        stage_wait();
        __syncthreads();

        // per-lane inputs: 4 features' xn hi/lo (bf16) and seg (u8)
        const bf16_t* xhp = xh + (size_t)row * IN_F + kf;
        const bf16_t* xlp = xl + (size_t)row * IN_F + kf;
        const uint32_t spack = *(const uint32_t*)(seg + (size_t)row * IN_F + kf);
        bf16_t xhv[4], xlv[4];
        int    sv[4];
        #pragma unroll
        for (int f = 0; f < 4; ++f) {
            xhv[f] = xhp[f];
            xlv[f] = xlp[f];
            sv[f]  = (int)((spack >> (8 * f)) & 0xFF);
        }

        // A fragments per ISA 16-bit A layout:
        //   lane-half 0: elems 0-7 -> K=0-7,  elems 8-15 -> K=16-23
        //   lane-half 1: elems 0-7 -> K=8-15, elems 8-15 -> K=24-31
        // expanded K = f_local*8 + s  =>  f_local = half + (e<8 ? 0 : 2), s = e&7
        v16bf afh, afl, ofrag;
        #pragma unroll
        for (int e = 0; e < 16; ++e) {
            const int f   = half + ((e < 8) ? 0 : 2);
            const int s   = e & 7;
            const bool hit = (sv[f] == s);
            afh[e]   = hit ? xhv[f] : (bf16_t)0.0f;
            afl[e]   = hit ? xlv[f] : (bf16_t)0.0f;
            ofrag[e] = hit ? (bf16_t)1.0f : (bf16_t)0.0f;
        }

        // B fragments: N = lane&15, element e -> K same as A (table order)
        #pragma unroll
        for (int nt = 0; nt < 4; ++nt) {
            v16bf fAh, fAl, fBh, fBl;
            #pragma unroll
            for (int e = 0; e < 16; ++e) {
                const int kk = (e < 8) ? (8 * half + e) : (16 + 8 * half + (e - 8));
                fAh[e] = lds_ah[kk][nt * 16 + m];
                fAl[e] = lds_al[kk][nt * 16 + m];
                fBh[e] = lds_bh[kk][nt * 16 + m];
                fBl[e] = lds_bl[kk][nt * 16 + m];
            }

            // a-term: xh*wh + xl*wh + xh*wl
            acc[nt] = __builtin_amdgcn_wmma_f32_16x16x32_bf16(
                false, afh, false, fAh, (short)0, acc[nt], false, false);
            acc[nt] = __builtin_amdgcn_wmma_f32_16x16x32_bf16(
                false, afl, false, fAh, (short)0, acc[nt], false, false);
            acc[nt] = __builtin_amdgcn_wmma_f32_16x16x32_bf16(
                false, afh, false, fAl, (short)0, acc[nt], false, false);
            // b-term: indicator (exact) * (wh + wl)
            acc[nt] = __builtin_amdgcn_wmma_f32_16x16x32_bf16(
                false, ofrag, false, fBh, (short)0, acc[nt], false, false);
            acc[nt] = __builtin_amdgcn_wmma_f32_16x16x32_bf16(
                false, ofrag, false, fBl, (short)0, acc[nt], false, false);
        }

        // prefetch next chunk of weights (global_prefetch_b8)
        if (kf + 4 < IN_F) {
            __builtin_prefetch(ah + (size_t)(kglob + KC) * OUT_F + cbase + kcol, 0, 1);
            __builtin_prefetch(bh + (size_t)(kglob + KC) * OUT_F + cbase + kcol, 0, 1);
        }
    }

    // C/D layout: VGPR r -> M = r + 8*half, N = lane&15
    #pragma unroll
    for (int nt = 0; nt < 4; ++nt) {
        #pragma unroll
        for (int r = 0; r < 8; ++r) {
            const int orow = rbase0 + wave * 16 + r + 8 * half;
            const int ocol = cbase + nt * 16 + m;
            y[(size_t)orow * OUT_F + ocol] = acc[nt][r];
        }
    }
}

// ---------------------------------------------------------------------------
extern "C" void kernel_launch(void* const* d_in, const int* in_sizes, int n_in,
                              void* d_out, int out_size, void* d_ws, size_t ws_size,
                              hipStream_t stream) {
    const float* x     = (const float*)d_in[0];
    const float* aW    = (const float*)d_in[1];
    const float* bW    = (const float*)d_in[2];
    const float* gamma = (const float*)d_in[3];
    const float* beta  = (const float*)d_in[4];
    float* y = (float*)d_out;

    const int rows = in_sizes[0] / IN_F;             // 2048
    const size_t nElem = (size_t)rows * IN_F;        // xn elements
    const size_t wElem = (size_t)IN_F * NSEG * OUT_F; // weight elements

    // ws layout: xh, xl (bf16), seg (u8), then 4 bf16 weight arrays
    char* p = (char*)d_ws;
    bf16_t*  xh_ws  = (bf16_t*)p;                 p += nElem * sizeof(bf16_t);
    bf16_t*  xl_ws  = (bf16_t*)p;                 p += nElem * sizeof(bf16_t);
    uint8_t* seg_ws = (uint8_t*)p;                p += nElem;
    p = (char*)(((uintptr_t)p + 255) & ~(uintptr_t)255);
    bf16_t* ah_ws = (bf16_t*)p;                   p += wElem * sizeof(bf16_t);
    bf16_t* al_ws = (bf16_t*)p;                   p += wElem * sizeof(bf16_t);
    bf16_t* bh_ws = (bf16_t*)p;                   p += wElem * sizeof(bf16_t);
    bf16_t* bl_ws = (bf16_t*)p;

    wsplit_kernel<<<(int)(wElem / (256 * 4)), 256, 0, stream>>>(
        aW, bW, ah_ws, al_ws, bh_ws, bl_ws);
    ln_seg_kernel<<<rows, IN_F, 0, stream>>>(x, gamma, beta, xh_ws, xl_ws, seg_ws);

    dim3 grid(OUT_F / BN, rows / BM);
    pwl_wmma_kernel<<<grid, 256, 0, stream>>>(
        xh_ws, xl_ws, seg_ws, ah_ws, al_ws, bh_ws, bl_ws, y);
}
